// ConditionedGAT_35854386987543
// MI455X (gfx1250) — compile-verified, matrix operand order checked by reading, not run on hardware
//
#include <hip/hip_runtime.h>

// ---------------- constants matching the reference setup ----------------
static constexpr int NN = 80000;   // num_nodes (not derivable from in_sizes)
static constexpr int DD = 64;      // feature dim
static constexpr int HH = 4;       // heads
// ------------------------------------------------------------------------

typedef __attribute__((ext_vector_type(16))) _Float16 v16h;
typedef __attribute__((ext_vector_type(8)))  float    v8f;

// ---- WMMA fragment loaders (16-bit A 16x32 layout, ISA 7.12.2) ----
__device__ __forceinline__ v16h frag_a(const _Float16* A, int lda, int row0, int k0, int lane) {
  v16h a;
  int m  = row0 + (lane & 15);
  int hi = (lane >> 4) & 1;
  const _Float16* p = A + m * lda + k0;
#pragma unroll
  for (int j = 0; j < 8; ++j) {
    int k = (j < 4) ? (hi ? 8 + 2 * j : 2 * j)
                    : (hi ? 24 + 2 * (j - 4) : 16 + 2 * (j - 4));
    a[2 * j]     = p[k];
    a[2 * j + 1] = p[k + 1];
  }
  return a;
}

__device__ __forceinline__ v16h frag_b(const _Float16* B, int ldb, int k0, int col0, int lane) {
  v16h b;
  int n  = col0 + (lane & 15);
  int hi = (lane >> 4) & 1;
#pragma unroll
  for (int j = 0; j < 8; ++j) {
    int k = (j < 4) ? (hi ? 8 + 2 * j : 2 * j)
                    : (hi ? 24 + 2 * (j - 4) : 16 + 2 * (j - 4));
    b[2 * j]     = B[(k0 + k) * ldb + n];
    b[2 * j + 1] = B[(k0 + k + 1) * ldb + n];
  }
  return b;
}

__device__ __forceinline__ float sigmoidf(float x) { return 1.0f / (1.0f + __expf(-x)); }

// ================= kernel 1: xl = (sigmoid(score) * hidden) @ W =================
__global__ void k_gemm_xl(const float* __restrict__ hidden, const float* __restrict__ score,
                          const float* __restrict__ W, float* __restrict__ xl) {
  __shared__ _Float16 As[64 * 64];
  __shared__ _Float16 Ws[64 * 64];
  int n0 = blockIdx.x * 64;
  int tid = threadIdx.x;
  for (int idx = tid; idx < 64 * 64; idx += 256) {
    int r = idx >> 6, c = idx & 63;
    float sg = sigmoidf(score[n0 + r]);
    As[idx] = (_Float16)(sg * hidden[(n0 + r) * 64 + c]);
    Ws[idx] = (_Float16)W[idx];
  }
  __syncthreads();
  int wave = tid >> 5, lane = tid & 31;
  for (int t = wave; t < 16; t += 8) {
    int row0 = (t >> 2) * 16, col0 = (t & 3) * 16;
    v8f acc = {};
#pragma unroll
    for (int ks = 0; ks < 2; ++ks) {
      v16h a = frag_a(As, 64, row0, ks * 32, lane);
      v16h b = frag_b(Ws, 64, ks * 32, col0, lane);
      acc = __builtin_amdgcn_wmma_f32_16x16x32_f16(false, a, false, b, (short)0, acc, false, false);
    }
    int n = lane & 15, mb = (lane >> 4) * 8;
#pragma unroll
    for (int r = 0; r < 8; ++r)
      xl[(n0 + row0 + mb + r) * 64 + col0 + n] = acc[r];
  }
}

// ================= kernel 2: attention coefficients =================
__global__ void k_att(const float* __restrict__ xl, const float* __restrict__ attS,
                      const float* __restrict__ attD, float* __restrict__ asrc,
                      float* __restrict__ adst) {
  int i = blockIdx.x * blockDim.x + threadIdx.x;
  if (i >= NN * HH) return;
  int n = i >> 2, h = i & 3;
  const float* x = xl + n * 64 + h * 16;
  float s = 0.f, d = 0.f;
#pragma unroll
  for (int c = 0; c < 16; ++c) {
    float xv = x[c];
    s += xv * attS[h * 16 + c];
    d += xv * attD[h * 16 + c];
  }
  asrc[i] = s;
  adst[i] = d;
}

// ================= kernel 3: edge pass 1 — alpha + segment max =================
__global__ void k_edge_alpha(const int* __restrict__ ei, int E,
                             const float* __restrict__ asrc, const float* __restrict__ adst,
                             float* __restrict__ alpha, unsigned* __restrict__ mkey) {
  int i = blockIdx.x * blockDim.x + threadIdx.x;
  if (i >= E * HH) return;
  int e = i >> 2, h = i & 3;
  int s = ei[e], d = ei[E + e];
  float a = asrc[s * 4 + h] + adst[d * 4 + h];
  a = (a >= 0.f) ? a : 0.2f * a;                 // leaky_relu(0.2)
  alpha[i] = a;
  unsigned u = __float_as_uint(a);
  u = (u & 0x80000000u) ? ~u : (u | 0x80000000u); // monotone key
  atomicMax(&mkey[d * 4 + h], u);
}

// ================= kernel 4: decode segment max =================
__global__ void k_decode_max(const unsigned* __restrict__ mkey, float* __restrict__ mval) {
  int i = blockIdx.x * blockDim.x + threadIdx.x;
  if (i >= NN * HH) return;
  unsigned u = mkey[i];
  u = (u & 0x80000000u) ? (u ^ 0x80000000u) : ~u;
  float m = __uint_as_float(u);
  if (!(fabsf(m) < 3.0e38f)) m = 0.f;            // non-finite -> 0 (reference rule)
  mval[i] = m;
}

// ================= kernel 5: edge pass 2 — exp + segment sum =================
__global__ void k_edge_exp(const int* __restrict__ ei, int E, float* __restrict__ alpha,
                           const float* __restrict__ mval, float* __restrict__ denom) {
  int i = blockIdx.x * blockDim.x + threadIdx.x;
  if (i >= E * HH) return;
  int e = i >> 2, h = i & 3;
  int d = ei[E + e];
  float ex = __expf(alpha[i] - mval[d * 4 + h]);
  alpha[i] = ex;
  atomicAdd(&denom[d * 4 + h], ex);
}

// ================= kernel 6: denom adjust =================
__global__ void k_denom_adj(float* __restrict__ denom) {
  int i = blockIdx.x * blockDim.x + threadIdx.x;
  if (i >= NN * HH) return;
  float v = denom[i];
  denom[i] = (v > 0.f) ? v : 1.f;
}

// ================= kernel 7: edge pass 3 — message aggregation (wave per edge) =========
__global__ void k_edge_msg(const int* __restrict__ ei, int E, const float* __restrict__ ex,
                           const float* __restrict__ den, const float* __restrict__ xl,
                           float* __restrict__ out) {
  int e = blockIdx.x * 8 + (threadIdx.x >> 5);
  if (e >= E) return;
  int lane = threadIdx.x & 31;
  int s = ei[e], d = ei[E + e];
  int c0 = lane * 2;
  int h = c0 >> 4;
  float w = ex[e * 4 + h] / den[d * 4 + h];
  const float2* px = reinterpret_cast<const float2*>(xl + s * 64);
  float2 xv = px[lane];
  atomicAdd(&out[d * 64 + c0], w * xv.x);
  atomicAdd(&out[d * 64 + c0 + 1], w * xv.y);
}

// ================= kernel 8: bias + LayerNorm (wave per node) =================
__global__ void k_ln(float* __restrict__ out, const float* __restrict__ bias,
                     const float* __restrict__ g, const float* __restrict__ b) {
  int n = blockIdx.x * 8 + (threadIdx.x >> 5);
  if (n >= NN) return;
  int lane = threadIdx.x & 31;
  int c0 = lane * 2;
  float o0 = out[n * 64 + c0] + bias[c0];
  float o1 = out[n * 64 + c0 + 1] + bias[c0 + 1];
  float s = o0 + o1;
  for (int off = 16; off; off >>= 1) s += __shfl_xor(s, off);
  float mu = s * (1.f / 64.f);
  float d0 = o0 - mu, d1 = o1 - mu;
  float v = d0 * d0 + d1 * d1;
  for (int off = 16; off; off >>= 1) v += __shfl_xor(v, off);
  float inv = rsqrtf(v * (1.f / 64.f) + 1e-5f);
  out[n * 64 + c0]     = d0 * inv * g[c0] + b[c0];
  out[n * 64 + c0 + 1] = d1 * inv * g[c0 + 1] + b[c0 + 1];
}

// ================= kernel 9: gate GEMM + hidden update =================
__global__ void k_gate(const float* __restrict__ outln, const float* __restrict__ relB, int rep,
                       const float* __restrict__ gateW, const float* __restrict__ gateB,
                       float* __restrict__ hidden) {
  __shared__ _Float16 As[64 * 128];
  __shared__ _Float16 Ws[128 * 64];
  int n0 = blockIdx.x * 64;
  int tid = threadIdx.x;
  for (int idx = tid; idx < 64 * 128; idx += 256) {
    int r = idx >> 7, c = idx & 127;
    float v = (c < 64) ? outln[(n0 + r) * 64 + c] : relB[((n0 + r) / rep) * 64 + (c - 64)];
    As[idx] = (_Float16)v;
    Ws[idx] = (_Float16)gateW[idx];
  }
  __syncthreads();
  int wave = tid >> 5, lane = tid & 31;
  for (int t = wave; t < 16; t += 8) {
    int row0 = (t >> 2) * 16, col0 = (t & 3) * 16;
    v8f acc = {};
#pragma unroll
    for (int ks = 0; ks < 4; ++ks) {
      v16h a = frag_a(As, 128, row0, ks * 32, lane);
      v16h b = frag_b(Ws, 64, ks * 32, col0, lane);
      acc = __builtin_amdgcn_wmma_f32_16x16x32_f16(false, a, false, b, (short)0, acc, false, false);
    }
    int n = lane & 15, mb = (lane >> 4) * 8;
#pragma unroll
    for (int r = 0; r < 8; ++r) {
      int row = n0 + row0 + mb + r, col = col0 + n;
      float gte = sigmoidf(acc[r] + gateB[col]);
      float o = outln[row * 64 + col];
      hidden[row * 64 + col] += o * gte;
    }
  }
}

// ================= kernel 10: fused score_fn (two chained WMMA GEMMs) =================
__global__ void k_score(const float* __restrict__ hidden, const float* __restrict__ relB, int rep,
                        const float* __restrict__ linW, const float* __restrict__ linB,
                        const float* __restrict__ W1, const float* __restrict__ b1,
                        const float* __restrict__ W2col, const float* __restrict__ b2,
                        float* __restrict__ score) {
  __shared__ _Float16 As[64 * 128];
  __shared__ _Float16 LWs[128 * 64];
  __shared__ _Float16 Xs[64 * 64];
  __shared__ _Float16 W1s[64 * 64];
  __shared__ float    ssc[64];
  int n0 = blockIdx.x * 64;
  int tid = threadIdx.x;
  for (int idx = tid; idx < 64 * 128; idx += 256) {
    int r = idx >> 7, c = idx & 127;
    float v = (c < 64) ? hidden[(n0 + r) * 64 + c] : relB[((n0 + r) / rep) * 64 + (c - 64)];
    As[idx]  = (_Float16)v;
    LWs[idx] = (_Float16)linW[idx];
  }
  for (int idx = tid; idx < 64 * 64; idx += 256) W1s[idx] = (_Float16)W1[idx];
  if (tid < 64) ssc[tid] = 0.f;
  __syncthreads();

  int wave = tid >> 5, lane = tid & 31;
  // GEMM1: heuristic = [hidden|rel] @ linW + linB ; x = hidden * heuristic
  for (int t = wave; t < 16; t += 8) {
    int row0 = (t >> 2) * 16, col0 = (t & 3) * 16;
    v8f acc = {};
#pragma unroll
    for (int ks = 0; ks < 4; ++ks) {
      v16h a = frag_a(As, 128, row0, ks * 32, lane);
      v16h b = frag_b(LWs, 64, ks * 32, col0, lane);
      acc = __builtin_amdgcn_wmma_f32_16x16x32_f16(false, a, false, b, (short)0, acc, false, false);
    }
    int n = lane & 15, mb = (lane >> 4) * 8;
#pragma unroll
    for (int r = 0; r < 8; ++r) {
      int row = row0 + mb + r, col = col0 + n;
      float heur = acc[r] + linB[col];
      float x = hidden[(n0 + row) * 64 + col] * heur;
      Xs[row * 64 + col] = (_Float16)x;
    }
  }
  __syncthreads();
  // GEMM2: h = relu(x @ W1 + b1) ; score = h @ W2col + b2
  for (int t = wave; t < 16; t += 8) {
    int row0 = (t >> 2) * 16, col0 = (t & 3) * 16;
    v8f acc = {};
#pragma unroll
    for (int ks = 0; ks < 2; ++ks) {
      v16h a = frag_a(Xs, 64, row0, ks * 32, lane);
      v16h b = frag_b(W1s, 64, ks * 32, col0, lane);
      acc = __builtin_amdgcn_wmma_f32_16x16x32_f16(false, a, false, b, (short)0, acc, false, false);
    }
    int n = lane & 15, mb = (lane >> 4) * 8;
#pragma unroll
    for (int r = 0; r < 8; ++r) {
      int row = row0 + mb + r, col = col0 + n;
      float hv = acc[r] + b1[col];
      hv = (hv > 0.f) ? hv : 0.f;
      atomicAdd(&ssc[row], hv * W2col[col]);
    }
  }
  __syncthreads();
  if (tid < 64) score[n0 + tid] = ssc[tid] + b2[0];
}

// ================= small scalar kernels =================
__global__ void k_gather_rel(const int* __restrict__ r_index, const float* __restrict__ table,
                             float* __restrict__ relB, int B) {
  int i = blockIdx.x * blockDim.x + threadIdx.x;
  if (i >= B * 64) return;
  int b = i >> 6, c = i & 63;
  relB[i] = table[r_index[b] * 64 + c];
}

__global__ void k_scatter_rows(const int* __restrict__ idx, const float* __restrict__ vals,
                               float* __restrict__ dstbuf, int B) {
  int i = blockIdx.x * blockDim.x + threadIdx.x;
  if (i >= B * 64) return;
  int b = i >> 6, c = i & 63;
  dstbuf[idx[b] * 64 + c] = vals[i];
}

__global__ void k_head_score(const float* __restrict__ headE, const float* __restrict__ relB,
                             const int* __restrict__ h_index,
                             const float* __restrict__ linW, const float* __restrict__ linB,
                             const float* __restrict__ W1, const float* __restrict__ b1,
                             const float* __restrict__ W2col, const float* __restrict__ b2,
                             float* __restrict__ score) {
  int b = blockIdx.x, j = threadIdx.x;  // 64 threads
  __shared__ float hb[64], rb[64], xs[64], hs[64];
  hb[j] = headE[b * 64 + j];
  rb[j] = relB[b * 64 + j];
  __syncthreads();
  float acc = linB[j];
  for (int k = 0; k < 64; ++k) acc += hb[k] * linW[k * 64 + j];
  for (int k = 0; k < 64; ++k) acc += rb[k] * linW[(64 + k) * 64 + j];
  xs[j] = hb[j] * acc;
  __syncthreads();
  float a2 = b1[j];
  for (int k = 0; k < 64; ++k) a2 += xs[k] * W1[k * 64 + j];
  a2 = (a2 > 0.f) ? a2 : 0.f;
  hs[j] = a2 * W2col[j];
  __syncthreads();
  if (j == 0) {
    float s = 0.f;
    for (int k = 0; k < 64; ++k) s += hs[k];
    score[h_index[b]] = s + b2[0];
  }
}

__global__ void k_gather_out(const int* __restrict__ t_index, const float* __restrict__ score,
                             float* __restrict__ outp, int B) {
  int i = blockIdx.x * blockDim.x + threadIdx.x;
  if (i < B) outp[i] = score[t_index[i]];
}

// =========================== host launcher ===========================
extern "C" void kernel_launch(void* const* d_in, const int* in_sizes, int n_in,
                              void* d_out, int out_size, void* d_ws, size_t ws_size,
                              hipStream_t stream) {
  const int* ei        = (const int*)d_in[0];
  const int* h_index   = (const int*)d_in[1];
  const int* t_index   = (const int*)d_in[2];
  const int* r_index   = (const int*)d_in[3];
  const int* all_index = (const int*)d_in[4];
  const float* headE   = (const float*)d_in[6];
  const float* stext   = (const float*)d_in[7];
  const float* reltab  = (const float*)d_in[8];
  const float* gatW    = (const float*)d_in[9];
  const float* attS    = (const float*)d_in[10];
  const float* attD    = (const float*)d_in[11];
  const float* gatB    = (const float*)d_in[12];
  const float* lng     = (const float*)d_in[13];
  const float* lnb     = (const float*)d_in[14];
  const float* gateW   = (const float*)d_in[15];
  const float* gateB   = (const float*)d_in[16];
  const float* linW    = (const float*)d_in[17];
  const float* linB    = (const float*)d_in[18];
  const float* W1      = (const float*)d_in[19];
  const float* b1      = (const float*)d_in[20];
  const float* W2col   = (const float*)d_in[21];
  const float* b2      = (const float*)d_in[22];

  const int E   = in_sizes[0] / 2;
  const int B   = in_sizes[1];
  const int L   = in_sizes[9] / (DD * DD);
  const int rep = NN / B;

  float* ws = (float*)d_ws;
  float*    hidden = ws;              ws += (size_t)NN * 64;
  float*    scoreb = ws;              ws += NN;
  float*    xl     = ws;              ws += (size_t)NN * 64;
  float*    outb   = ws;              ws += (size_t)NN * 64;
  float*    asrc   = ws;              ws += NN * 4;
  float*    adst   = ws;              ws += NN * 4;
  unsigned* mkey   = (unsigned*)ws;   ws += NN * 4;
  float*    mval   = ws;              ws += NN * 4;
  float*    denom  = ws;              ws += NN * 4;
  float*    relB   = ws;              ws += (size_t)B * 64;
  float*    alpha  = ws;              ws += (size_t)E * 4;

  // ---------- init ----------
  hipMemsetAsync(hidden, 0, (size_t)NN * 64 * sizeof(float), stream);
  hipMemsetAsync(scoreb, 0, (size_t)NN * sizeof(float), stream);
  k_gather_rel<<<(B * 64 + 255) / 256, 256, 0, stream>>>(r_index, reltab, relB, B);
  k_scatter_rows<<<(B * 64 + 255) / 256, 256, 0, stream>>>(all_index, stext, hidden, B);
  k_scatter_rows<<<(B * 64 + 255) / 256, 256, 0, stream>>>(h_index, headE, hidden, B);
  k_head_score<<<B, 64, 0, stream>>>(headE, relB, h_index, linW, linB, W1, b1, W2col, b2, scoreb);

  const int nb64  = NN / 64;                 // 1250
  const int ebH   = (E * HH + 255) / 256;    // edge*head grid
  const int nbH   = (NN * HH + 255) / 256;
  const int ebW   = (E + 7) / 8;             // wave-per-edge grid
  const int nbW   = (NN + 7) / 8;

  for (int i = 0; i < L; ++i) {
    const float* Wi  = gatW  + (size_t)i * 64 * 64;
    const float* aSi = attS  + (size_t)i * 64;
    const float* aDi = attD  + (size_t)i * 64;
    const float* bGi = gatB  + (size_t)i * 64;
    const float* gi  = lng   + (size_t)i * 64;
    const float* bi  = lnb   + (size_t)i * 64;
    const float* gWi = gateW + (size_t)i * 128 * 64;
    const float* gBi = gateB + (size_t)i * 64;

    k_gemm_xl<<<nb64, 256, 0, stream>>>(hidden, scoreb, Wi, xl);
    k_att<<<nbH, 256, 0, stream>>>(xl, aSi, aDi, asrc, adst);

    hipMemsetAsync(mkey,  0, (size_t)NN * 4 * sizeof(unsigned), stream);
    hipMemsetAsync(denom, 0, (size_t)NN * 4 * sizeof(float), stream);
    hipMemsetAsync(outb,  0, (size_t)NN * 64 * sizeof(float), stream);

    k_edge_alpha<<<ebH, 256, 0, stream>>>(ei, E, asrc, adst, alpha, mkey);
    k_decode_max<<<nbH, 256, 0, stream>>>(mkey, mval);
    k_edge_exp<<<ebH, 256, 0, stream>>>(ei, E, alpha, mval, denom);
    k_denom_adj<<<nbH, 256, 0, stream>>>(denom);
    k_edge_msg<<<ebW, 256, 0, stream>>>(ei, E, alpha, denom, xl, outb);

    k_ln<<<nbW, 256, 0, stream>>>(outb, bGi, gi, bi);
    k_gate<<<nb64, 256, 0, stream>>>(outb, relB, rep, gWi, gBi, hidden);
    k_score<<<nb64, 256, 0, stream>>>(hidden, relB, rep, linW, linB, W1, b1, W2col, b2, scoreb);
  }

  k_gather_out<<<(B + 255) / 256, 256, 0, stream>>>(t_index, scoreb, (float*)d_out, B);
}
